// MSNN_57174604644744
// MI455X (gfx1250) — matrix operand order repaired and checked by reading, not compiled, not run on hardware
//
#include <hip/hip_runtime.h>
#include <hip/hip_bf16.h>

// ---------------------------------------------------------------------------
// SNN (2-layer LIF) for MI455X / gfx1250, wave32 + WMMA.
//   Kernel 1: cur1 = x @ w1^T + b1      (fp32, v_wmma_f32_16x16x4_f32)
//   Kernel 2: fused 128-step recurrent loop, LDS-resident state per WGP,
//             layer-2 GEMM via v_wmma_f32_16x16x32_f16 (spikes are exact in f16)
// ---------------------------------------------------------------------------

typedef __attribute__((ext_vector_type(2)))  float    v2f;
typedef __attribute__((ext_vector_type(8)))  float    v8f;
typedef __attribute__((ext_vector_type(16))) _Float16 v16h;

#define BATCH 2048
#define NI    1024
#define NH    2048
#define NO    10
#define TSTEP 128

__device__ __constant__ float kBETA = 0.95f;
__device__ __constant__ float kTHR  = 1.0f;

// ---------------------------------------------------------------------------
// Kernel 1: cur1[B,NH] = x[B,NI] @ w1[NH,NI]^T + b1
// Block: 256 threads = 8 waves; each wave -> one 16x16 output tile, K = NI.
// Grid: (B/16, NH/128).  x M-tile (16 x 1024 fp32 = 64 KB) staged in LDS.
// ---------------------------------------------------------------------------
__global__ void __launch_bounds__(256) snn_cur1_gemm(const float* __restrict__ x,
                                                     const float* __restrict__ w1,
                                                     const float* __restrict__ b1,
                                                     float* __restrict__ cur1)
{
    extern __shared__ float lds[];                 // 16 x NI x-tile
    const int tid   = threadIdx.x;
    const int wave  = tid >> 5;
    const int lane  = tid & 31;
    const int nl    = lane & 15;                   // N (B/C/D) or M (A) index
    const int khalf = lane >> 4;

    const int m0 = blockIdx.x * 16;
    const int n0 = blockIdx.y * 128 + wave * 16;

    // Cooperative x-tile load -> LDS (float4)
    {
        const float4* src = (const float4*)(x + (size_t)m0 * NI);
        float4*       dst = (float4*)lds;
        for (int i = tid; i < 16 * NI / 4; i += 256) dst[i] = src[i];
    }
    __syncthreads();

    // A 16x4 f32 layout: lane (m=nl, khalf) holds K = 2*khalf, 2*khalf+1
    // B  4x16 f32 layout: lane (n=nl, khalf) holds K = 2*khalf, 2*khalf+1
    const float* xa  = lds + nl * NI;
    const float* w1n = w1 + (size_t)(n0 + nl) * NI;

    v8f acc = {};
    for (int kk = 0; kk < NI; kk += 4) {
        v2f a, b;
        a.x = xa[kk + 2 * khalf + 0];
        a.y = xa[kk + 2 * khalf + 1];
        b.x = w1n[kk + 2 * khalf + 0];
        b.y = w1n[kk + 2 * khalf + 1];
        acc = __builtin_amdgcn_wmma_f32_16x16x4_f32(false, a, false, b,
                                                    (short)0, acc, false, false);
    }

    // C/D 16x16 f32 layout: n = lane&15, VGPR v -> m = v + 8*(lane>=16)
    const float bias = b1[n0 + nl];
    for (int v = 0; v < 8; ++v) {
        const int m = v + 8 * khalf;
        cur1[(size_t)(m0 + m) * NH + (n0 + nl)] = acc[v] + bias;
    }
}

// ---------------------------------------------------------------------------
// Kernel 2: fused 128-step LIF recurrence. One block per 16 batch rows.
// LDS: cur1 tile (128 KB) + mem1 state (128 KB) + 8 KB reduction = 264 KB.
// 8 waves split the NH=2048 reduction (256 each = 8 f16 WMMAs/step/wave).
// ---------------------------------------------------------------------------
__global__ void __launch_bounds__(256) snn_loop(const float* __restrict__ cur1,
                                                const float* __restrict__ w2,
                                                const float* __restrict__ b2,
                                                float* __restrict__ spk2_rec,
                                                float* __restrict__ mem2_rec)
{
    extern __shared__ float lds[];
    float* s_cur1 = lds;                 // 16*NH fp32
    float* s_mem1 = lds + 16 * NH;       // 16*NH fp32
    float* s_red  = lds + 32 * NH;       // 8*256 fp32

    const int tid  = threadIdx.x;
    const int wave = tid >> 5;
    const int lane = tid & 31;
    const int nl   = lane & 15;          // output-unit col (B/C/D) and batch row (A)
    const int hh   = lane >> 4;          // half-wave selector
    const int m0   = blockIdx.x * 16;

    // Stage cur1 tile, zero mem1 state.
    {
        const float4* src = (const float4*)(cur1 + (size_t)m0 * NH);
        float4*       dst = (float4*)s_cur1;
        for (int i = tid; i < 16 * NH / 4; i += 256) dst[i] = src[i];
        float4 z = {0.f, 0.f, 0.f, 0.f};
        float4* mz = (float4*)s_mem1;
        for (int i = tid; i < 16 * NH / 4; i += 256) mz[i] = z;
    }

    // Resident w2 B-fragments (f16), this wave's K-slice [k0w, k0w+256).
    // B 32x16 f16 operand, contiguous k order (same as A): lane (n=nl, hh) element j holds K = 16*hh + j.
    const int k0w = wave * 256;
    v16h Bfrag[8];
    for (int f = 0; f < 8; ++f) {
        const int kb = k0w + f * 32 + 16 * hh;
        for (int j = 0; j < 16; ++j) {
            const float wv = (nl < NO) ? w2[(size_t)nl * NH + (kb + j)] : 0.0f;
            Bfrag[f][j] = (_Float16)wv;
        }
    }
    const float bias2 = (nl < NO) ? b2[nl] : 0.0f;

    v8f mem2 = {};
    __syncthreads();

    for (int t = 0; t < TSTEP; ++t) {
        // (1) mem1 LIF update (reset-by-subtraction), 128 elems / thread.
        for (int i = tid; i < 16 * NH; i += 256) {
            const float m = s_mem1[i];
            const float r = (m > kTHR) ? kTHR : 0.0f;
            s_mem1[i] = kBETA * m + s_cur1[i] - r;
        }
        __syncthreads();

        // (2) spk1 A-fragments built from LDS on the fly, 8 WMMAs over K=256.
        // A 16x32 f16 operand, contiguous k order (same as B): lane m = nl; element j -> K = 16*hh + j.
        v8f acc = {};
        const float* mrow = s_mem1 + nl * NH;
        for (int f = 0; f < 8; ++f) {
            const int kbase = k0w + f * 32;
            v16h A;
            for (int j = 0; j < 16; ++j) {
                const int k = 16 * hh + j;
                const float m = mrow[kbase + k];
                A[j] = (m > kTHR) ? (_Float16)1.0f : (_Float16)0.0f;
            }
            acc = __builtin_amdgcn_wmma_f32_16x16x32_f16(false, A, false, Bfrag[f],
                                                         (short)0, acc, false, false);
        }

        // (3) 8-wave reduction of the 16x16 partial cur2 tile through LDS.
        for (int v = 0; v < 8; ++v) s_red[wave * 256 + v * 32 + lane] = acc[v];
        __syncthreads();
        v8f cur2;
        for (int v = 0; v < 8; ++v) {
            float s = 0.0f;
            for (int w = 0; w < 8; ++w) s += s_red[w * 256 + v * 32 + lane];
            cur2[v] = s + bias2;
        }

        // (4) mem2 LIF update in C-layout registers (redundant per wave,
        //     identical values -> deterministic).
        v8f spk2;
        for (int v = 0; v < 8; ++v) {
            const float mprev = mem2[v];
            const float r     = (mprev > kTHR) ? kTHR : 0.0f;
            const float m     = kBETA * mprev + cur2[v] - r;
            mem2[v] = m;
            spk2[v] = (m - kTHR > 0.0f) ? 1.0f : 0.0f;
        }

        // (5) wave 0 records this step: [T, B, NO] for spk2 and mem2.
        if (wave == 0 && nl < NO) {
            const size_t base = ((size_t)t * BATCH + m0) * NO + nl;
            for (int v = 0; v < 8; ++v) {
                const int m = v + 8 * hh;
                spk2_rec[base + (size_t)m * NO] = spk2[v];
                mem2_rec[base + (size_t)m * NO] = mem2[v];
            }
        }
        __syncthreads();   // guard s_red / s_mem1 reuse next step
    }
}

// ---------------------------------------------------------------------------
extern "C" void kernel_launch(void* const* d_in, const int* in_sizes, int n_in,
                              void* d_out, int out_size, void* d_ws, size_t ws_size,
                              hipStream_t stream)
{
    (void)in_sizes; (void)n_in; (void)out_size; (void)ws_size;

    const float* x  = (const float*)d_in[0];
    const float* w1 = (const float*)d_in[1];
    const float* b1 = (const float*)d_in[2];
    const float* w2 = (const float*)d_in[3];
    const float* b2 = (const float*)d_in[4];

    float* out      = (float*)d_out;
    float* spk2_rec = out;                                   // [T,B,NO]
    float* mem2_rec = out + (size_t)TSTEP * BATCH * NO;      // [T,B,NO]
    float* cur1     = (float*)d_ws;                          // [B,NH] = 16 MB

    // Kernel 1: loop-invariant layer-1 currents.
    {
        dim3 grid(BATCH / 16, NH / 128);
        const size_t shmem = (size_t)16 * NI * sizeof(float);          // 64 KB
        snn_cur1_gemm<<<grid, 256, shmem, stream>>>(x, w1, b1, cur1);
    }

    // Kernel 2: fused 128-step recurrence, LDS-resident per workgroup.
    {
        dim3 grid(BATCH / 16);
        const size_t shmem = (size_t)(32 * NH + 8 * 256) * sizeof(float); // 264 KB
        snn_loop<<<grid, 256, shmem, stream>>>(cur1, w2, b2, spk2_rec, mem2_rec);
    }
}
